// RelGraphConvLayer_43705587204167
// MI455X (gfx1250) — compile-verified, matrix operand order checked by reading, not run on hardware
//
#include <hip/hip_runtime.h>

// ---------------------------------------------------------------------------
// RelGraphConv (R-GCN) layer for MI455X / gfx1250
//   N=20000 nodes, IN=OUT=512, R=8 relations, B=4 bases, E=80000 edges/rel
//   f16 WMMA GEMMs (v_wmma_f32_16x16x32_f16, f32 accum) + f32 atomic scatter.
//   GEMM: block = 8 waves, 128 rows x 64 cols per block; each wave holds 4
//   16x16 f32 accumulators and reuses its A fragment across 4 WMMAs.
// ---------------------------------------------------------------------------

#define NN    20000
#define INF   512
#define OUTF  512
#define RREL  8
#define BBAS  4
#define EDG   80000

#define LDS_STRIDE 520          // halves; 1040B rows -> 4-bank stagger per col
#define LDS_BYTES  (64 * LDS_STRIDE * 2)   // 66,560 B (dynamic LDS)

typedef __attribute__((ext_vector_type(16))) _Float16 v16h;
typedef __attribute__((ext_vector_type(8)))  _Float16 v8h;
typedef __attribute__((ext_vector_type(8)))  float    v8f;

// ---------------------------------------------------------------------------
// K1: build 9 f16 weight matrices: W9[0]=loop_weight, W9[1+r]=sum_b w_comp*bases
// ---------------------------------------------------------------------------
__global__ __launch_bounds__(256) void build_weights(
    const float* __restrict__ w_comp,      // [R,B]
    const float* __restrict__ bases,       // [B,IN,OUT]
    const float* __restrict__ loop_weight, // [IN,OUT]
    _Float16* __restrict__ W9)             // [9,IN,OUT] f16
{
    int tid = blockIdx.x * blockDim.x + threadIdx.x;
    const int per = INF * OUTF;
    if (tid >= 9 * per) return;
    int m  = tid / per;
    int io = tid - m * per;
    float v;
    if (m == 0) {
        v = loop_weight[io];
    } else {
        int r = m - 1;
        v = 0.0f;
        #pragma unroll
        for (int b = 0; b < BBAS; ++b)
            v = fmaf(w_comp[r * BBAS + b], bases[(size_t)b * per + io], v);
    }
    W9[tid] = (_Float16)v;
}

// ---------------------------------------------------------------------------
// K2: convert x fp32 -> f16 (vectorized x4)
// ---------------------------------------------------------------------------
__global__ __launch_bounds__(256) void convert_x(
    const float* __restrict__ x, _Float16* __restrict__ xh)
{
    int tid = blockIdx.x * blockDim.x + threadIdx.x;
    const int total4 = (NN * INF) / 4;
    if (tid >= total4) return;
    float4 f = ((const float4*)x)[tid];
    _Float16* p = xh + (size_t)tid * 4;
    p[0] = (_Float16)f.x; p[1] = (_Float16)f.y;
    p[2] = (_Float16)f.z; p[3] = (_Float16)f.w;
}

// ---------------------------------------------------------------------------
// K3/K4: per-relation in-degree (float); clamp applied at use site
// ---------------------------------------------------------------------------
__global__ __launch_bounds__(256) void zero_deg(float* __restrict__ deg)
{
    int tid = blockIdx.x * blockDim.x + threadIdx.x;
    if (tid < RREL * NN) deg[tid] = 0.0f;
}

__global__ __launch_bounds__(256) void acc_deg(
    const int* __restrict__ dst, float* __restrict__ deg)
{
    int tid = blockIdx.x * blockDim.x + threadIdx.x;
    if (tid >= RREL * EDG) return;
    int r = tid / EDG;
    int d = dst[tid];                    // dst[r*E + e]
    unsafeAtomicAdd(&deg[r * NN + d], 1.0f);
}

// ---------------------------------------------------------------------------
// K5: WMMA GEMM  C[M,N] = A[M,512] * W[512,N]   (A,W f16; C f32)
//   grid (ceil(1250/8), 8); block 256 = 8 wave32s.
//   Block stages W[:, n0..n0+63] into LDS transposed/padded (dyn LDS 66.6KB);
//   each wave: one 16-row tile x four 16-col tiles, A fragment reused 4x.
//   Fragment layouts per CDNA5 ISA 7.12.2 (wave32):
//     A: lane<16 row M=lane&15, K chunks {0..7,16..23}; lane>=16 same rows,
//        K chunks {8..15,24..31}  -> two global_load_b128 per k-step.
//     B: lane&15 = column, same K split  -> two ds_load_b128 per fragment.
//     C: element v -> row (v + 8*(lane>>4)), col lane&15.
// ---------------------------------------------------------------------------
__global__ __launch_bounds__(256) void gemm_f16_wmma(
    const _Float16* __restrict__ A,   // [NN, 512] f16
    const _Float16* __restrict__ W,   // [512, 512] f16 (one matrix)
    float* __restrict__ C)            // [NN, 512] f32
{
    extern __shared__ _Float16 WldsT[];   // [64][LDS_STRIDE] col-major panel

    const int n0 = blockIdx.y * 64;
    // cooperative stage: coalesced b128 global reads, transposed LDS writes
    for (int t = threadIdx.x; t < 512 * 64 / 8; t += 256) {
        int k  = t >> 3;
        int c8 = (t & 7) * 8;
        v8h v = *(const v8h*)(W + (size_t)k * OUTF + n0 + c8);
        #pragma unroll
        for (int j = 0; j < 8; ++j)
            WldsT[(c8 + j) * LDS_STRIDE + k] = v[j];
    }
    __syncthreads();

    const int wave  = threadIdx.x >> 5;
    const int lane  = threadIdx.x & 31;
    const int mtile = blockIdx.x * 8 + wave;
    if (mtile >= NN / 16) return;          // wave-uniform: EXEC stays all-1s

    const int row   = mtile * 16 + (lane & 15);
    const int khalf = lane >> 4;
    const _Float16* arow  = A + (size_t)row * INF + khalf * 8;
    const _Float16* bbase = WldsT + (lane & 15) * LDS_STRIDE + khalf * 8;

    v8f c0 = {}, c1 = {}, c2 = {}, c3 = {};
    #pragma unroll 2
    for (int k0 = 0; k0 < INF; k0 += 32) {
        __builtin_prefetch(arow + k0 + 64, 0, 1);      // global_prefetch_b8
        v8h alo = *(const v8h*)(arow + k0);
        v8h ahi = *(const v8h*)(arow + k0 + 16);
        v16h a = __builtin_shufflevector(alo, ahi,
                 0,1,2,3,4,5,6,7,8,9,10,11,12,13,14,15);

        #pragma unroll
        for (int j = 0; j < 4; ++j) {
            const _Float16* bp = bbase + (size_t)j * 16 * LDS_STRIDE + k0;
            v8h blo = *(const v8h*)(bp);
            v8h bhi = *(const v8h*)(bp + 16);
            v16h b = __builtin_shufflevector(blo, bhi,
                     0,1,2,3,4,5,6,7,8,9,10,11,12,13,14,15);
            v8f& acc = (j == 0) ? c0 : (j == 1) ? c1 : (j == 2) ? c2 : c3;
            acc = __builtin_amdgcn_wmma_f32_16x16x32_f16(
                    /*neg_a=*/false, a, /*neg_b=*/false, b,
                    /*c_mod=*/(short)0, acc, /*reuse_a=*/false, /*reuse_b=*/false);
        }
    }

    float* crow = C + (size_t)(mtile * 16 + khalf * 8) * OUTF + n0 + (lane & 15);
    #pragma unroll
    for (int j = 0; j < 4; ++j) {
        const v8f& acc = (j == 0) ? c0 : (j == 1) ? c1 : (j == 2) ? c2 : c3;
        #pragma unroll
        for (int v = 0; v < 8; ++v)
            crow[(size_t)v * OUTF + j * 16] = acc[v];
    }
}

// ---------------------------------------------------------------------------
// K6: per-relation scatter: out[dst[e]] += h[src[e]] / max(deg[dst[e]],1)
//   one thread per (edge, 4 cols); native global_atomic_add_f32.
// ---------------------------------------------------------------------------
__global__ __launch_bounds__(256) void scatter_edges(
    const float* __restrict__ h,    // [NN,512]
    const int*   __restrict__ src,  // [E] (this relation)
    const int*   __restrict__ dst,  // [E]
    const float* __restrict__ deg,  // [NN] (this relation)
    float* __restrict__ out)        // [NN,512]
{
    int tid = blockIdx.x * blockDim.x + threadIdx.x;
    if (tid >= EDG * 128) return;
    int e  = tid >> 7;
    int c4 = (tid & 127) * 4;
    int s = src[e];
    int d = dst[e];
    float dg  = deg[d];
    float inv = 1.0f / (dg > 1.0f ? dg : 1.0f);
    float4 hv = *(const float4*)(h + (size_t)s * OUTF + c4);
    float* o  = out + (size_t)d * OUTF + c4;
    unsafeAtomicAdd(o + 0, hv.x * inv);
    unsafeAtomicAdd(o + 1, hv.y * inv);
    unsafeAtomicAdd(o + 2, hv.z * inv);
    unsafeAtomicAdd(o + 3, hv.w * inv);
}

// ---------------------------------------------------------------------------
// K7: out = relu(out + h_bias + sum_r rel_bias[r])   (broadcast biases)
// ---------------------------------------------------------------------------
__global__ __launch_bounds__(256) void finalize(
    float* __restrict__ out,
    const float* __restrict__ rel_bias,  // [R,OUT]
    const float* __restrict__ h_bias)    // [OUT]
{
    int tid = blockIdx.x * blockDim.x + threadIdx.x;
    if (tid >= NN * 128) return;
    int c4 = (tid & 127) * 4;
    float4 o = ((float4*)out)[tid];
    float b0 = h_bias[c4 + 0], b1 = h_bias[c4 + 1],
          b2 = h_bias[c4 + 2], b3 = h_bias[c4 + 3];
    #pragma unroll
    for (int r = 0; r < RREL; ++r) {
        const float* rb = rel_bias + r * OUTF + c4;
        b0 += rb[0]; b1 += rb[1]; b2 += rb[2]; b3 += rb[3];
    }
    o.x = fmaxf(o.x + b0, 0.0f);
    o.y = fmaxf(o.y + b1, 0.0f);
    o.z = fmaxf(o.z + b2, 0.0f);
    o.w = fmaxf(o.w + b3, 0.0f);
    ((float4*)out)[tid] = o;
}

// ---------------------------------------------------------------------------
// Launch
// ---------------------------------------------------------------------------
extern "C" void kernel_launch(void* const* d_in, const int* in_sizes, int n_in,
                              void* d_out, int out_size, void* d_ws, size_t ws_size,
                              hipStream_t stream) {
    const float* x           = (const float*)d_in[0];
    const int*   src         = (const int*)  d_in[1];
    const int*   dst         = (const int*)  d_in[2];
    const float* w_comp      = (const float*)d_in[3];
    const float* bases       = (const float*)d_in[4];
    const float* rel_bias    = (const float*)d_in[5];
    const float* loop_weight = (const float*)d_in[6];
    const float* h_bias      = (const float*)d_in[7];
    float* out = (float*)d_out;

    // workspace layout (all offsets 16B aligned)
    char* ws = (char*)d_ws;
    _Float16* xh  = (_Float16*)(ws);                               // 20,480,000 B
    _Float16* W9  = (_Float16*)(ws + 20480000);                    //  4,718,592 B
    float*    deg = (float*)   (ws + 20480000 + 4718592);          //    640,000 B
    float*    hbf = (float*)   (ws + 20480000 + 4718592 + 640000); // 40,960,000 B

    // 1) basis-combined weights (f16) ; 2) x -> f16
    build_weights<<<(9 * INF * OUTF + 255) / 256, 256, 0, stream>>>(
        w_comp, bases, loop_weight, W9);
    convert_x<<<((NN * INF / 4) + 255) / 256, 256, 0, stream>>>(x, xh);

    // 3) per-relation degrees
    zero_deg<<<(RREL * NN + 255) / 256, 256, 0, stream>>>(deg);
    acc_deg<<<(RREL * EDG + 255) / 256, 256, 0, stream>>>(dst, deg);

    dim3 ggrid((NN / 16 + 7) / 8, OUTF / 64, 1);  // (157, 8)

    // 4) self-loop GEMM writes d_out fully
    gemm_f16_wmma<<<ggrid, 256, LDS_BYTES, stream>>>(xh, W9, out);

    // 5) per-relation GEMM -> scatter (stream-serialized, h buffer reused)
    for (int r = 0; r < RREL; ++r) {
        gemm_f16_wmma<<<ggrid, 256, LDS_BYTES, stream>>>(
            xh, W9 + (size_t)(1 + r) * INF * OUTF, hbf);
        scatter_edges<<<(EDG * 128 + 255) / 256, 256, 0, stream>>>(
            hbf, src + r * EDG, dst + r * EDG, deg + r * NN, out);
    }

    // 6) biases + ReLU
    finalize<<<(NN * 128 + 255) / 256, 256, 0, stream>>>(out, rel_bias, h_bias);
}
